// ReadoutLayer_80384607912339
// MI455X (gfx1250) — compile-verified, matrix operand order checked by reading, not run on hardware
//
#include <hip/hip_runtime.h>
#include <math.h>

// ---------------- problem constants (from reference) ----------------
#define DIN    512
#define NH     512
#define NB     32
#define NT     2000
#define NM     (NB * NT)        // 64000 rows of flattened Wx
#define MTILE  128
#define NTILE  64
#define NMT    (NM / MTILE)     // 500 M-tiles
#define SEGLEN 100
#define NSEG   (NT / SEGLEN)    // 20 segments

#define ALPHA_LO 0.8187307530779818f
#define ALPHA_HI 0.9607894391523232f
#define BN_EPS   1e-5f

// LDS W-tile row stride in dwords (bf16 pairs): 256 data + 4 pad.
// Row stride = 260 dwords -> 260 % 64 = 4, so the 16 lanes of a b128
// fragment read (4 dwords each, rows r..r+15) cover all 64 banks once.
#define WROW 260

typedef __attribute__((ext_vector_type(16))) __bf16 v16bf;
typedef __attribute__((ext_vector_type(8)))  float  v8f;
typedef __attribute__((ext_vector_type(4)))  float  v4f;
typedef __attribute__((ext_vector_type(2)))  float  v2f;
typedef __attribute__((ext_vector_type(4)))  unsigned int v4u;

union FragBF { v16bf v; unsigned int u[8]; v4u q[2]; };
union FragF  { v8f   v; float        f[8]; };

// f32 pair -> packed bf16 dword.  Hardware packed convert (RNE); the clang
// builtin isn't declared for gfx1250 on this toolchain, so issue the
// instruction directly.
__device__ __forceinline__ unsigned int pack_bf16(float lo, float hi) {
    unsigned int r;
    asm("v_cvt_pk_bf16_f32 %0, %1, %2" : "=v"(r) : "v"(lo), "v"(hi));
    return r;
}

// Build a 16x32 bf16 A fragment from an f32 row pointer (global).
// ISA layout (16-bit A 16x32): lanes 0-15 hold K = {k0..k0+7, k0+16..k0+23},
// lanes 16-31 hold K = {k0+8..k0+15, k0+24..k0+31}.
__device__ __forceinline__ void load_frag_a(FragBF& fr, const float* __restrict__ row,
                                            int k0, int half) {
    const int ka = k0 + half * 8;
    const int kb = k0 + 16 + half * 8;
    v4f a0 = *(const v4f*)(row + ka);
    v4f a1 = *(const v4f*)(row + ka + 4);
    v4f a2 = *(const v4f*)(row + kb);
    v4f a3 = *(const v4f*)(row + kb + 4);
    fr.u[0] = pack_bf16(a0.x, a0.y);
    fr.u[1] = pack_bf16(a0.z, a0.w);
    fr.u[2] = pack_bf16(a1.x, a1.y);
    fr.u[3] = pack_bf16(a1.z, a1.w);
    fr.u[4] = pack_bf16(a2.x, a2.y);
    fr.u[5] = pack_bf16(a2.z, a2.w);
    fr.u[6] = pack_bf16(a3.x, a3.y);
    fr.u[7] = pack_bf16(a3.z, a3.w);
}

// B fragment: two ds_load_b128 from the pre-converted LDS W tile.
__device__ __forceinline__ void load_frag_b_lds(FragBF& fr, const unsigned int* wl,
                                                int rowl, int k0, int half) {
    const int ka = k0 + half * 8;          // element index, /2 = dword index
    const int kb = k0 + 16 + half * 8;
    fr.q[0] = *(const v4u*)(wl + rowl * WROW + (ka >> 1));
    fr.q[1] = *(const v4u*)(wl + rowl * WROW + (kb >> 1));
}

// ---------------------------------------------------------------------------
// Kernel 1: Wx = x @ W^T + b  (bf16 WMMA, f32 accum) + per-tile BN partials.
// Grid: (NMT, NH/NTILE). Block: 256 (8 waves). Wave w owns rows
// [m0+16w, m0+16w+16) x [n0, n0+64) as four 16x16 WMMA accumulators.
// The 64x512 W tile is converted to bf16 once into LDS by the whole block.
// ---------------------------------------------------------------------------
__global__ __launch_bounds__(256)
void gemm_bn_stats_kernel(const float* __restrict__ x, const float* __restrict__ W,
                          const float* __restrict__ bias, float* __restrict__ WX,
                          float* __restrict__ psum, float* __restrict__ psq) {
    __shared__ unsigned int Wlds[64 * WROW];       // 64 rows of 512 bf16 (+pad)

    const int tid  = threadIdx.x;
    const int wave = tid >> 5;
    const int lane = tid & 31;
    const int half = lane >> 4;
    const int l15  = lane & 15;
    const int m0 = blockIdx.x * MTILE + wave * 16;
    const int n0 = blockIdx.y * NTILE;

    // ---- cooperative convert of W tile: 64 rows x 256 f32-pairs ----
    {
        const float* Wt = W + (size_t)n0 * DIN;
        for (int p = tid; p < 64 * 256; p += 256) {
            const int row = p >> 8;                // 0..63
            const int cp  = p & 255;               // pair index in row
            v2f pr = *(const v2f*)(Wt + (size_t)row * DIN + cp * 2);
            Wlds[row * WROW + cp] = pack_bf16(pr.x, pr.y);
        }
    }
    __syncthreads();

    v8f acc0 = {}; v8f acc1 = {}; v8f acc2 = {}; v8f acc3 = {};
    const float* arow = x + (size_t)(m0 + l15) * DIN;

#pragma unroll 2
    for (int k0 = 0; k0 < DIN; k0 += 32) {
        FragBF af, bf0, bf1, bf2, bf3;
        load_frag_a(af, arow, k0, half);
        load_frag_b_lds(bf0, Wlds,  0 + l15, k0, half);
        load_frag_b_lds(bf1, Wlds, 16 + l15, k0, half);
        load_frag_b_lds(bf2, Wlds, 32 + l15, k0, half);
        load_frag_b_lds(bf3, Wlds, 48 + l15, k0, half);
        acc0 = __builtin_amdgcn_wmma_f32_16x16x32_bf16(false, af.v, false, bf0.v,
                                                       (short)0, acc0, false, false);
        acc1 = __builtin_amdgcn_wmma_f32_16x16x32_bf16(false, af.v, false, bf1.v,
                                                       (short)0, acc1, false, false);
        acc2 = __builtin_amdgcn_wmma_f32_16x16x32_bf16(false, af.v, false, bf2.v,
                                                       (short)0, acc2, false, false);
        acc3 = __builtin_amdgcn_wmma_f32_16x16x32_bf16(false, af.v, false, bf3.v,
                                                       (short)0, acc3, false, false);
    }

    // Epilogue: add bias, store Wx, reduce per-column partial sum / sum-sq.
    // C/D layout: VGPR r => row m0 + r + 8*half; col = n0 + 16*sub + l15.
    __shared__ float lsum[8][4][16];
    __shared__ float lsq [8][4][16];

    v8f accs[4] = {acc0, acc1, acc2, acc3};
#pragma unroll
    for (int s = 0; s < 4; ++s) {
        const int col = n0 + s * 16 + l15;
        const float bv = bias[col];
        FragF cc; cc.v = accs[s];
        float sum = 0.f, sq = 0.f;
#pragma unroll
        for (int r = 0; r < 8; ++r) {
            const int row = m0 + r + half * 8;
            const float v = cc.f[r] + bv;
            WX[(size_t)row * NH + col] = v;
            sum += v;
            sq  += v * v;
        }
        // lanes l and l+16 hold the same column (different rows)
        sum += __shfl_xor(sum, 16, 32);
        sq  += __shfl_xor(sq,  16, 32);
        if (half == 0) { lsum[wave][s][l15] = sum; lsq[wave][s][l15] = sq; }
    }
    __syncthreads();
    if (tid < 64) {
        const int s = tid >> 4, c = tid & 15;
        float sum = 0.f, sq = 0.f;
#pragma unroll
        for (int w = 0; w < 8; ++w) { sum += lsum[w][s][c]; sq += lsq[w][s][c]; }
        const int col = n0 + s * 16 + c;
        psum[(size_t)blockIdx.x * NH + col] = sum;
        psq [(size_t)blockIdx.x * NH + col] = sq;
    }
}

// ---------------------------------------------------------------------------
// Kernel 2: finalize BN (deterministic fixed-order reduction over 500 tile
// partials), fold gamma/beta into scale/shift, clip alpha, precompute a^L.
// ---------------------------------------------------------------------------
__global__ void bn_finalize_kernel(const float* __restrict__ psum, const float* __restrict__ psq,
                                   const float* __restrict__ gamma, const float* __restrict__ beta,
                                   const float* __restrict__ alpha,
                                   float* __restrict__ scale, float* __restrict__ shift,
                                   float* __restrict__ aclip, float* __restrict__ pal) {
    const int h = blockIdx.x * blockDim.x + threadIdx.x;
    if (h >= NH) return;
    float s = 0.f, q = 0.f;
    for (int mt = 0; mt < NMT; ++mt) {
        s += psum[(size_t)mt * NH + h];
        q += psq [(size_t)mt * NH + h];
    }
    const float inv_n = 1.0f / (float)NM;
    const float mean  = s * inv_n;
    const float var   = q * inv_n - mean * mean;
    const float rstd  = rsqrtf(var + BN_EPS);
    const float sc    = gamma[h] * rstd;
    scale[h] = sc;
    shift[h] = beta[h] - mean * sc;
    const float a = fminf(fmaxf(alpha[h], ALPHA_LO), ALPHA_HI);
    aclip[h] = a;
    pal[h]   = __powf(a, (float)SEGLEN);
}

// ---------------------------------------------------------------------------
// Kernel 3a: per-segment zero-start recurrence: s_seg = sum_i a^{L-1-i}(1-a)z_i
// One thread per (b, seg, h). Coalesced over h; prefetch ahead in t.
// ---------------------------------------------------------------------------
__global__ __launch_bounds__(256)
void seg_scan_kernel(const float* __restrict__ WX, const float* __restrict__ scale,
                     const float* __restrict__ shift, const float* __restrict__ aclip,
                     float* __restrict__ sseg) {
    const int g   = blockIdx.x * 256 + threadIdx.x;   // [0, NB*NSEG*NH)
    const int h   = g % NH;
    const int seg = (g / NH) % NSEG;
    const int b   = g / (NH * NSEG);
    const float a = aclip[h], om = 1.0f - a;
    const float sc = scale[h], sh = shift[h];
    const float* p = WX + ((size_t)(b * NT + seg * SEGLEN)) * NH + h;
    float u = 0.f;
#pragma unroll 4
    for (int i = 0; i < SEGLEN; ++i) {
        if (i + 16 < SEGLEN)
            __builtin_prefetch((const void*)(p + (size_t)(i + 16) * NH), 0, 3);
        const float z = sc * p[(size_t)i * NH] + sh;
        u = a * u + om * z;
    }
    sseg[g] = u;
}

// ---------------------------------------------------------------------------
// Kernel 3b: chain segment boundary states: u_{seg+1} = a^L * u_seg + s_seg.
// ---------------------------------------------------------------------------
__global__ void seg_chain_kernel(const float* __restrict__ sseg, const float* __restrict__ ut0,
                                 const float* __restrict__ pal, float* __restrict__ ustart) {
    const int g = blockIdx.x * blockDim.x + threadIdx.x;  // [0, NB*NH)
    const int h = g % NH;
    const int b = g / NH;
    const float p = pal[h];
    float u = ut0[g];
    for (int seg = 0; seg < NSEG; ++seg) {
        const size_t idx = ((size_t)(b * NSEG + seg)) * NH + h;
        ustart[idx] = u;
        u = p * u + sseg[idx];
    }
}

// ---------------------------------------------------------------------------
// Kernel 4: re-run the recurrence inside each segment with the correct start
// state and accumulate softmax(ut).  One 512-thread block per (b, seg).
// Next step's Wx load is software-pipelined past the barrier reductions.
// ---------------------------------------------------------------------------
__global__ __launch_bounds__(512)
void softmax_acc_kernel(const float* __restrict__ WX, const float* __restrict__ scale,
                        const float* __restrict__ shift, const float* __restrict__ aclip,
                        const float* __restrict__ ustart, float* __restrict__ accseg) {
    __shared__ float red[17];
    const int h    = threadIdx.x;           // 0..511
    const int seg  = blockIdx.x % NSEG;
    const int b    = blockIdx.x / NSEG;
    const int wid  = h >> 5;
    const int lane = h & 31;
    const float a  = aclip[h], om = 1.0f - a;
    const float sc = scale[h], sh = shift[h];
    const size_t sidx = ((size_t)(b * NSEG + seg)) * NH + h;
    float u = ustart[sidx];
    const float* p = WX + ((size_t)(b * NT + seg * SEGLEN)) * NH + h;
    float acc = 0.f;

    float zraw = p[0];                       // pipelined load for i=0
    for (int i = 0; i < SEGLEN; ++i) {
        const float zcur = zraw;
        if (i + 1 < SEGLEN) zraw = p[(size_t)(i + 1) * NH];   // overlap with reductions

        const float z = sc * zcur + sh;
        u = a * u + om * z;

        // ---- block max over 512 channels ----
        float m = u;
        for (int o = 16; o > 0; o >>= 1) m = fmaxf(m, __shfl_xor(m, o, 32));
        if (lane == 0) red[wid] = m;
        __syncthreads();
        if (wid == 0) {
            float t = (lane < 16) ? red[lane] : -INFINITY;
            for (int o = 8; o > 0; o >>= 1) t = fmaxf(t, __shfl_xor(t, o, 32));
            if (lane == 0) red[16] = t;
        }
        __syncthreads();
        m = red[16];

        const float e = __expf(u - m);

        // ---- block sum ----
        float s = e;
        for (int o = 16; o > 0; o >>= 1) s += __shfl_xor(s, o, 32);
        if (lane == 0) red[wid] = s;
        __syncthreads();
        if (wid == 0) {
            float t = (lane < 16) ? red[lane] : 0.f;
            for (int o = 8; o > 0; o >>= 1) t += __shfl_xor(t, o, 32);
            if (lane == 0) red[16] = t;
        }
        __syncthreads();
        s = red[16];

        acc += e / s;
        __syncthreads();   // protect red[] before next iteration
    }
    accseg[sidx] = acc;
}

// ---------------------------------------------------------------------------
// Kernel 5: deterministic fixed-order sum of per-segment accumulators.
// ---------------------------------------------------------------------------
__global__ void final_sum_kernel(const float* __restrict__ accseg, float* __restrict__ out) {
    const int g = blockIdx.x * blockDim.x + threadIdx.x;  // [0, NB*NH)
    const int h = g % NH;
    const int b = g / NH;
    float s = 0.f;
    for (int seg = 0; seg < NSEG; ++seg)
        s += accseg[((size_t)(b * NSEG + seg)) * NH + h];
    out[g] = s;   // out is [B, 1, H] contiguous
}

// ---------------------------------------------------------------------------
// Host-side launch
// ---------------------------------------------------------------------------
extern "C" void kernel_launch(void* const* d_in, const int* in_sizes, int n_in,
                              void* d_out, int out_size, void* d_ws, size_t ws_size,
                              hipStream_t stream) {
    const float* x     = (const float*)d_in[0];
    const float* W     = (const float*)d_in[1];
    const float* bias  = (const float*)d_in[2];
    const float* alpha = (const float*)d_in[3];
    const float* gamma = (const float*)d_in[4];
    const float* beta  = (const float*)d_in[5];
    const float* ut0   = (const float*)d_in[6];
    float* out = (float*)d_out;

    // workspace layout (floats)
    float* ws = (float*)d_ws;
    const size_t OFF_WX     = 0;                                    // 64000*512
    const size_t OFF_PSUM   = OFF_WX   + (size_t)NM * NH;           // 500*512
    const size_t OFF_PSQ    = OFF_PSUM + (size_t)NMT * NH;
    const size_t OFF_SCALE  = OFF_PSQ  + (size_t)NMT * NH;          // 512
    const size_t OFF_SHIFT  = OFF_SCALE + NH;
    const size_t OFF_ACLIP  = OFF_SHIFT + NH;
    const size_t OFF_PAL    = OFF_ACLIP + NH;
    const size_t OFF_SSEG   = OFF_PAL   + NH;                       // 32*20*512
    const size_t OFF_USTART = OFF_SSEG   + (size_t)NB * NSEG * NH;
    const size_t OFF_ACCSEG = OFF_USTART + (size_t)NB * NSEG * NH;

    float* WX     = ws + OFF_WX;
    float* psum   = ws + OFF_PSUM;
    float* psq    = ws + OFF_PSQ;
    float* scale  = ws + OFF_SCALE;
    float* shift  = ws + OFF_SHIFT;
    float* aclip  = ws + OFF_ACLIP;
    float* pal    = ws + OFF_PAL;
    float* sseg   = ws + OFF_SSEG;
    float* ustart = ws + OFF_USTART;
    float* accseg = ws + OFF_ACCSEG;

    // 1) GEMM + BN partial stats
    gemm_bn_stats_kernel<<<dim3(NMT, NH / NTILE), 256, 0, stream>>>(x, W, bias, WX, psum, psq);
    // 2) BN finalize + alpha prep
    bn_finalize_kernel<<<dim3((NH + 255) / 256), 256, 0, stream>>>(psum, psq, gamma, beta, alpha,
                                                                   scale, shift, aclip, pal);
    // 3a) per-segment zero-start recurrence
    seg_scan_kernel<<<dim3((NB * NSEG * NH) / 256), 256, 0, stream>>>(WX, scale, shift, aclip, sseg);
    // 3b) chain segment boundary states
    seg_chain_kernel<<<dim3((NB * NH) / 256), 256, 0, stream>>>(sseg, ut0, pal, ustart);
    // 4) softmax accumulate per (b, seg)
    softmax_acc_kernel<<<dim3(NB * NSEG), 512, 0, stream>>>(WX, scale, shift, aclip, ustart, accseg);
    // 5) deterministic final reduction
    final_sum_kernel<<<dim3((NB * NH) / 256), 256, 0, stream>>>(accseg, out);
}
